// MolNet_Layer_55113020342588
// MI455X (gfx1250) — matrix operand order repaired and checked by reading, not compiled, not run on hardware
//
#include <hip/hip_runtime.h>
#include <math.h>

#define NNODES 40000
#define NEDGES 640000
#define CH 64
#define NG 50

typedef _Float16 half16 __attribute__((ext_vector_type(16)));
typedef _Float16 half8  __attribute__((ext_vector_type(8)));
typedef float    f32x8  __attribute__((ext_vector_type(8)));

// Swizzled-weight LDS offsets (halfs): S2S,V2S(192K),SL,S2V,V2V,VL
#define OFF_S2S 0
#define OFF_V2S 8192
#define OFF_SL  20480
#define OFF_S2V 28672
#define OFF_V2V 36864
#define OFF_VL  45056
#define W_TOTAL_HALFS 53248   // 106,496 bytes
#define W_CHUNKS 6656         // 16-byte chunks

// ---------------------------------------------------------------- helpers ---
__device__ __forceinline__ f32x8 wmma_f16(half16 a, half16 b, f32x8 c) {
  // v_wmma_f32_16x16x32_f16: D = A(16x32 f16) * B(32x16 f16) + C(16x16 f32)
  return __builtin_amdgcn_wmma_f32_16x16x32_f16(false, a, false, b, (short)0, c,
                                                false, false);
}

__device__ __forceinline__ half16 make_a(half8 lo, half8 hi) {
  half16 r;
#pragma unroll
  for (int i = 0; i < 8; ++i) { r[i] = lo[i]; r[8 + i] = hi[i]; }
  return r;
}

// 8 consecutive f32 -> 8 f16 (32B aligned source)
__device__ __forceinline__ half8 run8_from_f32(const float* __restrict__ p) {
  float4 a = *(const float4*)p;
  float4 b = *(const float4*)(p + 4);
  half8 r;
  r[0] = (_Float16)a.x; r[1] = (_Float16)a.y; r[2] = (_Float16)a.z; r[3] = (_Float16)a.w;
  r[4] = (_Float16)b.x; r[5] = (_Float16)b.y; r[6] = (_Float16)b.z; r[7] = (_Float16)b.w;
  return r;
}

// ||v||_2 over the 3-axis for 8 consecutive channels of one node row [3][64]
__device__ __forceinline__ half8 vn_run8(const float* __restrict__ vrow, int c0) {
  half8 r;
#pragma unroll
  for (int i = 0; i < 8; ++i) {
    float x = vrow[c0 + i];
    float y = vrow[64 + c0 + i];
    float z = vrow[128 + c0 + i];
    r[i] = (_Float16)sqrtf(x * x + y * y + z * z);
  }
  return r;
}

// GaussianSmearing(0, 5, 50) for 8 consecutive gaussians (zero-padded past 50)
__device__ __forceinline__ half8 rbf_run8(int q0, float dist, float coeff) {
  half8 r;
#pragma unroll
  for (int i = 0; i < 8; ++i) {
    int q = q0 + i;
    float d = dist - (float)q * (5.0f / 49.0f);
    float v = __expf(coeff * d * d);
    r[i] = (q < NG) ? (_Float16)v : (_Float16)0.0f;
  }
  return r;
}

// ---------------------------------------------------------------- kernels ---

// zero output accumulators + degree buffer (deterministic per launch)
__global__ void zero_kernel(float* __restrict__ out, float* __restrict__ deg,
                            long nOut, int nDeg) {
  long i = (long)blockIdx.x * blockDim.x + threadIdx.x;
  long stride = (long)gridDim.x * blockDim.x;
  for (long j = i; j < nOut; j += stride) out[j] = 0.0f;
  for (long j = i; j < nDeg; j += stride) deg[j] = 0.0f;
}

// deg[c] = segment_sum(edge_attr by col)
__global__ void deg_kernel(const int* __restrict__ ei,
                           const float* __restrict__ ea,
                           float* __restrict__ deg) {
  int e = blockIdx.x * blockDim.x + threadIdx.x;
  if (e < NEDGES) atomicAdd(&deg[ei[NEDGES + e]], ea[e]);
}

__global__ void dis_kernel(const float* __restrict__ deg, float* __restrict__ dis) {
  int n = blockIdx.x * blockDim.x + threadIdx.x;
  if (n < NNODES) {
    float d = deg[n];
    dis[n] = d > 0.0f ? rsqrtf(d) : 0.0f;
  }
}

// Re-swizzle a [Ksrc,64] fp32 weight into WMMA B-fragment f16 layout (Kpad rows).
// Fragment f = kstep*4 + ntile; within frag: lane(32) x 16 halfs.
// B lane layout (ISA 7.12.2): lane = n%16 + 16*g, half h -> K = 32*kstep + 8*g + h + (h>=8 ? 8 : 0).
__global__ void prep_w(const float* __restrict__ src, _Float16* __restrict__ dst,
                       int Ksrc, int Kpad) {
  int d = blockIdx.x * blockDim.x + threadIdx.x;
  int total = Kpad * 64;
  if (d >= total) return;
  int f = d >> 9;            // 512 halfs per fragment
  int w = d & 511;
  int lane = w >> 4, h = w & 15;
  int ks = f >> 2, nt = f & 3;
  int gg = lane >> 4, nl = lane & 15;
  int k = ks * 32 + 8 * gg + h + (h >= 8 ? 8 : 0);
  int n = nt * 16 + nl;
  float v = (k < Ksrc) ? src[(long)k * 64 + n] : 0.0f;
  dst[d] = (_Float16)v;
}

// Main per-edge message kernel: one wave handles 16 edges, 6 WMMA GEMMs.
// All six swizzled weight matrices are staged once per workgroup into LDS via
// async global->LDS copies (ASYNCcnt) and shared by the 8 waves.
__global__ __launch_bounds__(256) void molnet_edge_kernel(
    const float* __restrict__ scalar, const float* __restrict__ vec,
    const float* __restrict__ pos, const int* __restrict__ ei,
    const float* __restrict__ eattr, const float* __restrict__ disv,
    const _Float16* __restrict__ wAll,  // contiguous swizzled weights (106,496 B)
    const float* __restrict__ bS2S, const float* __restrict__ bV2S,
    const float* __restrict__ bSL, const float* __restrict__ bS2V,
    float* __restrict__ aggS, float* __restrict__ aggV) {
  __shared__ __attribute__((aligned(32))) _Float16 lw[W_TOTAL_HALFS];  // 104 KB
  __shared__ _Float16 stage[8][16 * 128];  // per-wave A restage [16 rows][128 K]
  __shared__ _Float16 svst[8][16 * 64];    // tanh(s@Ws2v+b) per-wave
  __shared__ float nrmS[8][16];
  __shared__ float relS[8][3][16];
  __shared__ int colS[8][16];

  const int lane = threadIdx.x & 31;
  const int wv = threadIdx.x >> 5;
  const int tile = blockIdx.x * 8 + wv;
  const int g = lane >> 4;   // K half-group for A, lane group for D rows
  const int ml = lane & 15;  // this lane's edge row (A layout)
  const int nl = lane & 15;  // this lane's column (C/D layout)
  const int mg = g * 8;      // D-row base for this lane group
  const int e = tile * 16 + ml;

  // ---- async-stage all weights into LDS (overlapped with edge setup) ----
  {
    const unsigned lds_base = (unsigned)(size_t)(const void*)&lw[0];
    const char* gsrc = (const char*)wAll;
#pragma unroll 1
    for (int it = 0; it < W_CHUNKS / 256; ++it) {   // 26 iters * 256 thr = 6656
      int idx = it * 256 + (int)threadIdx.x;
      unsigned lds_off = lds_base + (unsigned)idx * 16u;
      const char* ga = gsrc + (size_t)idx * 16;
      asm volatile("global_load_async_to_lds_b128 %0, %1, off"
                   :: "v"(lds_off), "v"(ga)
                   : "memory");
    }
  }

  const int nr = ei[e];           // row  (source j)
  const int nc = ei[NEDGES + e];  // col  (target i)

  const float px = pos[(long)nc * 3 + 0] - pos[(long)nr * 3 + 0];
  const float py = pos[(long)nc * 3 + 1] - pos[(long)nr * 3 + 1];
  const float pz = pos[(long)nc * 3 + 2] - pos[(long)nr * 3 + 2];
  const float dist = sqrtf(px * px + py * py + pz * pz);
  const float nrm = disv[nr] * disv[nc] * eattr[e];

  if (g == 0) {
    nrmS[wv][ml] = nrm;
    relS[wv][0][ml] = px;
    relS[wv][1][ml] = py;
    relS[wv][2][ml] = pz;
    colS[wv][ml] = nc;
  }
  // wait for async weight staging + make all LDS visible block-wide
  asm volatile("s_wait_asynccnt 0x0" ::: "memory");
  __syncthreads();

  // B fragment fetch from LDS (ds_load_b128 x2, contiguous 32B per lane)
  auto load_b = [&](int woff, int frag) -> half16 {
    return *(const half16*)&lw[woff + (frag * 32 + lane) * 16];
  };

  const float* si = scalar + (long)nc * CH;
  const float* sj = scalar + (long)nr * CH;
  const float* vi = vec + (long)nc * 3 * CH;
  const float* vj = vec + (long)nr * 3 * CH;

  // ---- GEMM1 (ss) + GEMM4 (sv pre-act): A=[si,sj], K=128, shared A ----
  f32x8 ssD[4] = {};
  f32x8 svD[4] = {};
#pragma unroll
  for (int ks = 0; ks < 4; ++ks) {
    const int b0 = ks * 32 + 8 * g;
    const int b1 = b0 + 16;
    half8 lo = run8_from_f32(b0 < 64 ? si + b0 : sj + (b0 - 64));
    half8 hi = run8_from_f32(b1 < 64 ? si + b1 : sj + (b1 - 64));
    half16 a = make_a(lo, hi);
#pragma unroll
    for (int nt = 0; nt < 4; ++nt) {
      ssD[nt] = wmma_f16(a, load_b(OFF_S2S, ks * 4 + nt), ssD[nt]);
      svD[nt] = wmma_f16(a, load_b(OFF_S2V, ks * 4 + nt), svD[nt]);
    }
  }
#pragma unroll
  for (int nt = 0; nt < 4; ++nt) {
    const int n = nt * 16 + nl;
    const float b1v = bS2S[n];
    const float b2v = bS2V[n];
#pragma unroll
    for (int r = 0; r < 8; ++r) {
      float v = ssD[nt][r] + b1v;
      v = v / (1.0f + __expf(-v));  // silu
      stage[wv][(mg + r) * 128 + n] = (_Float16)v;
      svst[wv][(mg + r) * 64 + n] = (_Float16)tanhf(svD[nt][r] + b2v);
    }
  }

  // ---- GEMM2 (vs): A=[|vi|,|vj|,rbf,pad], K=192 ----
  const float coeff = -0.5f / ((5.0f / 49.0f) * (5.0f / 49.0f));
  f32x8 vsD[4] = {};
#pragma unroll
  for (int ks = 0; ks < 6; ++ks) {
    const int b0 = ks * 32 + 8 * g;
    const int b1 = b0 + 16;
    half8 lo = (b0 < 64)    ? vn_run8(vi, b0)
               : (b0 < 128) ? vn_run8(vj, b0 - 64)
                            : rbf_run8(b0 - 128, dist, coeff);
    half8 hi = (b1 < 64)    ? vn_run8(vi, b1)
               : (b1 < 128) ? vn_run8(vj, b1 - 64)
                            : rbf_run8(b1 - 128, dist, coeff);
    half16 a = make_a(lo, hi);
#pragma unroll
    for (int nt = 0; nt < 4; ++nt)
      vsD[nt] = wmma_f16(a, load_b(OFF_V2S, ks * 4 + nt), vsD[nt]);
  }
#pragma unroll
  for (int nt = 0; nt < 4; ++nt) {
    const int n = nt * 16 + nl;
    const float bb = bV2S[n];
#pragma unroll
    for (int r = 0; r < 8; ++r) {
      float v = vsD[nt][r] + bb;
      v = v / (1.0f + __expf(-v));
      stage[wv][(mg + r) * 128 + 64 + n] = (_Float16)v;
    }
  }

  __builtin_amdgcn_wave_barrier();
  asm volatile("s_wait_dscnt 0" ::: "memory");

  // ---- GEMM3 (msg_s): A=[ss,vs] from LDS stage ----
  f32x8 msD[4] = {};
#pragma unroll
  for (int ks = 0; ks < 4; ++ks) {
    const int b0 = ks * 32 + 8 * g;
    const int b1 = b0 + 16;
    half8 lo = *(const half8*)&stage[wv][ml * 128 + b0];
    half8 hi = *(const half8*)&stage[wv][ml * 128 + b1];
    half16 a = make_a(lo, hi);
#pragma unroll
    for (int nt = 0; nt < 4; ++nt)
      msD[nt] = wmma_f16(a, load_b(OFF_SL, ks * 4 + nt), msD[nt]);
  }
#pragma unroll
  for (int nt = 0; nt < 4; ++nt) {
    const int n = nt * 16 + nl;
    const float bb = bSL[n];
#pragma unroll
    for (int r = 0; r < 8; ++r) {
      const int m = mg + r;
      float v = (msD[nt][r] + bb) * nrmS[wv][m];
      atomicAdd(&aggS[(long)colS[wv][m] * CH + n], v);
    }
  }

  // ---- vector path per axis: GEMM5 (vv) + GEMM6 (msg_v) ----
#pragma unroll 1
  for (int a3 = 0; a3 < 3; ++a3) {
    f32x8 vvD[4] = {};
#pragma unroll
    for (int ks = 0; ks < 4; ++ks) {
      const int b0 = ks * 32 + 8 * g;
      const int b1 = b0 + 16;
      half8 lo = run8_from_f32(b0 < 64 ? vi + a3 * 64 + b0 : vj + a3 * 64 + (b0 - 64));
      half8 hi = run8_from_f32(b1 < 64 ? vi + a3 * 64 + b1 : vj + a3 * 64 + (b1 - 64));
      half16 a = make_a(lo, hi);
#pragma unroll
      for (int nt = 0; nt < 4; ++nt)
        vvD[nt] = wmma_f16(a, load_b(OFF_V2V, ks * 4 + nt), vvD[nt]);
    }
    __builtin_amdgcn_wave_barrier();
    // restage [vv, sv*rel] as next A
#pragma unroll
    for (int nt = 0; nt < 4; ++nt) {
      const int n = nt * 16 + nl;
#pragma unroll
      for (int r = 0; r < 8; ++r)
        stage[wv][(mg + r) * 128 + n] = (_Float16)vvD[nt][r];
    }
#pragma unroll
    for (int it = 0; it < 32; ++it) {
      int idx = it * 32 + lane;  // 16*64 elements
      int m = idx >> 6;
      int n = idx & 63;
      float s = (float)svst[wv][idx];
      stage[wv][m * 128 + 64 + n] = (_Float16)(s * relS[wv][a3][m]);
    }
    __builtin_amdgcn_wave_barrier();
    asm volatile("s_wait_dscnt 0" ::: "memory");

    f32x8 mvD[4] = {};
#pragma unroll
    for (int ks = 0; ks < 4; ++ks) {
      const int b0 = ks * 32 + 8 * g;
      const int b1 = b0 + 16;
      half8 lo = *(const half8*)&stage[wv][ml * 128 + b0];
      half8 hi = *(const half8*)&stage[wv][ml * 128 + b1];
      half16 a = make_a(lo, hi);
#pragma unroll
      for (int nt = 0; nt < 4; ++nt)
        mvD[nt] = wmma_f16(a, load_b(OFF_VL, ks * 4 + nt), mvD[nt]);
    }
#pragma unroll
    for (int nt = 0; nt < 4; ++nt) {
      const int n = nt * 16 + nl;
#pragma unroll
      for (int r = 0; r < 8; ++r) {
        const int m = mg + r;
        float v = mvD[nt][r] * nrmS[wv][m];
        atomicAdd(&aggV[(long)colS[wv][m] * 192 + a3 * 64 + n], v);
      }
    }
  }
}

// out[0:N*64]   = silu(agg_s) + scalar
// out[N*64 : ] += vector
__global__ void finalize_kernel(float* __restrict__ out,
                                const float* __restrict__ scalar,
                                const float* __restrict__ vec) {
  long i = (long)blockIdx.x * blockDim.x + threadIdx.x;
  const long nS = (long)NNODES * 64;
  const long nV = (long)NNODES * 192;
  if (i < nS) {
    float a = out[i];
    out[i] = a / (1.0f + __expf(-a)) + scalar[i];
  } else if (i < nS + nV) {
    out[i] += vec[i - nS];
  }
}

// ---------------------------------------------------------------- launch ---
extern "C" void kernel_launch(void* const* d_in, const int* in_sizes, int n_in,
                              void* d_out, int out_size, void* d_ws, size_t ws_size,
                              hipStream_t stream) {
  (void)in_sizes; (void)n_in; (void)out_size; (void)ws_size;
  const float* scalar = (const float*)d_in[0];
  const float* vec    = (const float*)d_in[1];
  const float* pos    = (const float*)d_in[2];
  const int*   ei     = (const int*)d_in[3];
  const float* eattr  = (const float*)d_in[4];
  const float* Ws2s = (const float*)d_in[5];
  const float* bs2s = (const float*)d_in[6];
  const float* Wv2s = (const float*)d_in[7];
  const float* bv2s = (const float*)d_in[8];
  const float* Wsl  = (const float*)d_in[9];
  const float* bsl  = (const float*)d_in[10];
  const float* Ws2v = (const float*)d_in[11];
  const float* bs2v = (const float*)d_in[12];
  const float* Wv2v = (const float*)d_in[13];
  const float* Wvl  = (const float*)d_in[14];
  float* out = (float*)d_out;

  char* ws = (char*)d_ws;
  size_t off = 0;
  float* deg = (float*)(ws + off); off += (size_t)NNODES * 4;
  float* dis = (float*)(ws + off); off += (size_t)NNODES * 4;
  off = (off + 255) & ~(size_t)255;
  // contiguous swizzled weight block (order must match OFF_* LDS offsets)
  _Float16* wAll = (_Float16*)(ws + off);
  _Float16* wS2S = wAll + OFF_S2S;
  _Float16* wV2S = wAll + OFF_V2S;
  _Float16* wSL  = wAll + OFF_SL;
  _Float16* wS2V = wAll + OFF_S2V;
  _Float16* wV2V = wAll + OFF_V2V;
  _Float16* wVL  = wAll + OFF_VL;

  float* aggS = out;
  float* aggV = out + (size_t)NNODES * 64;
  const long nOut = (long)NNODES * 256;  // 64 + 192 per node

  zero_kernel<<<2048, 256, 0, stream>>>(out, deg, nOut, NNODES);

  prep_w<<<(128 * 64 + 255) / 256, 256, 0, stream>>>(Ws2s, wS2S, 128, 128);
  prep_w<<<(192 * 64 + 255) / 256, 256, 0, stream>>>(Wv2s, wV2S, 178, 192);
  prep_w<<<(128 * 64 + 255) / 256, 256, 0, stream>>>(Wsl,  wSL,  128, 128);
  prep_w<<<(128 * 64 + 255) / 256, 256, 0, stream>>>(Ws2v, wS2V, 128, 128);
  prep_w<<<(128 * 64 + 255) / 256, 256, 0, stream>>>(Wv2v, wV2V, 128, 128);
  prep_w<<<(128 * 64 + 255) / 256, 256, 0, stream>>>(Wvl,  wVL,  128, 128);

  deg_kernel<<<NEDGES / 256, 256, 0, stream>>>(ei, eattr, deg);
  dis_kernel<<<(NNODES + 255) / 256, 256, 0, stream>>>(deg, dis);

  // 40000 tiles of 16 edges, 8 waves per 256-thread block
  molnet_edge_kernel<<<5000, 256, 0, stream>>>(
      scalar, vec, pos, ei, eattr, dis, wAll,
      bs2s, bv2s, bsl, bs2v, aggS, aggV);

  finalize_kernel<<<(int)((nOut + 255) / 256), 256, 0, stream>>>(out, scalar, vec);
}
